// EncodeSegmentTime_81106162418455
// MI455X (gfx1250) — matrix operand order repaired and checked by reading, not compiled, not run on hardware
//
#include <hip/hip_runtime.h>
#include <hip/hip_bf16.h>
#include <math.h>

typedef _Float16 half_t;
typedef __attribute__((ext_vector_type(16))) _Float16 v16h;
typedef __attribute__((ext_vector_type(8)))  _Float16 v8h;
typedef __attribute__((ext_vector_type(8)))  float    v8f;
typedef __attribute__((ext_vector_type(4)))  unsigned int u32x4;
typedef __attribute__((ext_vector_type(8)))  int      i32x8;
typedef __attribute__((ext_vector_type(4)))  int      i32x4;

#define FREQ   24
#define SEG    21
#define HEADS  2
#define DH     12
#define MAXD   4
#define NSEQ   (64 * 512)      // 32768 sequences
#define KPAD   512             // padded K for the final GEMM (504 -> 512)
#define NUNITS 512
#define NROWS  32              // rows per block in the WMMA GEMM

// ---------------------------------------------------------------------------
// TDM: issue a 2D tile load (tile_h rows x tile_w f16 elems) Global -> LDS.
// Descriptor per CDNA5 ISA §8.3/8.4. Uniform args; wave-level DMA op.
// This toolchain uses the 6-arg builtin (clang-23 / therock-10.0 headers).
// ---------------------------------------------------------------------------
__device__ __forceinline__ void tdm_load_tile_f16(
    unsigned lds_off, const half_t* gaddr,
    unsigned tile_w, unsigned tile_h, unsigned row_stride_elems)
{
    unsigned long long ga = (unsigned long long)(uintptr_t)gaddr;
    u32x4 g0;
    g0[0] = 1u;                                          // count=1, user-mode
    g0[1] = lds_off;                                     // lds_addr (bytes)
    g0[2] = (unsigned)(ga & 0xFFFFFFFFu);                // global_addr[31:0]
    g0[3] = (unsigned)((ga >> 32) & 0x1FFFFFFu)          // global_addr[56:32]
          | 0x80000000u;                                 // type=2 ("image")
    i32x8 g1;
    g1[0] = 0x00010000;                                  // data_size=1 (2B), no flags
    g1[1] = (int)((row_stride_elems & 0xFFFFu) << 16);   // tensor_dim0 lo (use stride)
    g1[2] = (int)(((row_stride_elems >> 16) & 0xFFFFu)   // tensor_dim0 hi
          | ((0x100000u & 0xFFFFu) << 16));              // tensor_dim1 lo (1<<20 -> 0)
    g1[3] = (int)(((0x100000u >> 16) & 0xFFFFu)          // tensor_dim1 hi
          | ((tile_w & 0xFFFFu) << 16));                 // tile_dim0
    g1[4] = (int)(tile_h & 0xFFFFu);                     // tile_dim1 | tile_dim2=0
    g1[5] = (int)row_stride_elems;                       // tensor_dim0_stride lo
    g1[6] = 0;                                           // stride hi | dim1_stride lo
    g1[7] = 0;
    i32x4 g2 = {0, 0, 0, 0};
    i32x4 g3 = {0, 0, 0, 0};
    i32x8 g4 = {0, 0, 0, 0, 0, 0, 0, 0};                 // unused trailing group
    __builtin_amdgcn_tensor_load_to_lds(g0, g1, g2, g3, g4, 0);
}

// ---------------------------------------------------------------------------
// Kernel 0: transpose + pad wd [504,512] f32 -> wdT [512 (n-major)][512 (k)] f16
// ---------------------------------------------------------------------------
__global__ __launch_bounds__(256) void convert_wd_kernel(
    const float* __restrict__ wd, half_t* __restrict__ wdT)
{
    int i = blockIdx.x * 256 + threadIdx.x;          // over 512*512
    int u = i >> 9;          // output unit (N dim)
    int k = i & 511;         // padded K
    float v = (k < FREQ * SEG) ? wd[k * NUNITS + u] : 0.0f;
    wdT[(size_t)u * KPAD + k] = (half_t)v;
}

// ---------------------------------------------------------------------------
// Kernel 1: fused posenc + QKV + 2-head rel-pos attention + LN + FF + LN.
// One wave32 per sequence; thread s (<21) owns query row s.
// ---------------------------------------------------------------------------
__global__ __launch_bounds__(32) void attn_ff_kernel(
    const float* __restrict__ inp,
    const float* __restrict__ wq, const float* __restrict__ bq,
    const float* __restrict__ wk, const float* __restrict__ bk,
    const float* __restrict__ wv, const float* __restrict__ bv,
    const float* __restrict__ relk, const float* __restrict__ relv,
    const float* __restrict__ lag,  const float* __restrict__ lab,
    const float* __restrict__ w1,   const float* __restrict__ b1,
    const float* __restrict__ w2,   const float* __restrict__ b2,
    const float* __restrict__ lfg,  const float* __restrict__ lfb,
    half_t* __restrict__ ffbuf)
{
    __shared__ float xs[SEG][FREQ];
    __shared__ float qs[SEG][FREQ];
    __shared__ float ks[SEG][FREQ];
    __shared__ float vs[SEG][FREQ];
    __shared__ float as[SEG][FREQ];
    __shared__ float srk[2 * MAXD + 1][DH];
    __shared__ float srv[2 * MAXD + 1][DH];

    const int n = blockIdx.x;
    const int s = threadIdx.x;

    for (int i = s; i < (2 * MAXD + 1) * DH; i += 32) {
        srk[i / DH][i % DH] = relk[i];
        srv[i / DH][i % DH] = relv[i];
    }

    if (s < SEG) {
        for (int f = 0; f < FREQ; ++f) {
            float expo  = (float)(2 * (f >> 1)) / (float)FREQ;
            float denom = __powf(10000.0f, expo);
            float ang   = (float)s / denom;
            float pe    = (f & 1) ? __cosf(ang) : __sinf(ang);
            xs[s][f] = inp[(size_t)n * (FREQ * SEG) + f * SEG + s] + pe * 0.01f + 0.01f;
        }
        for (int f = 0; f < FREQ; ++f) {
            float aq = bq[f], ak = bk[f], av = bv[f];
            for (int j = 0; j < FREQ; ++j) {
                float xv = xs[s][j];
                aq = fmaf(xv, wq[j * FREQ + f], aq);
                ak = fmaf(xv, wk[j * FREQ + f], ak);
                av = fmaf(xv, wv[j * FREQ + f], av);
            }
            qs[s][f] = fmaxf(aq, 0.0f);
            ks[s][f] = fmaxf(ak, 0.0f);
            vs[s][f] = fmaxf(av, 0.0f);
        }
    }
    __syncthreads();

    if (s < SEG) {
        float o[FREQ];
        const float scale = rsqrtf((float)DH);
        #pragma unroll
        for (int h = 0; h < HEADS; ++h) {
            float logits[SEG];
            float mx = -1e30f;
            #pragma unroll
            for (int kq = 0; kq < SEG; ++kq) {
                int id = kq - s;
                id = id < -MAXD ? -MAXD : (id > MAXD ? MAXD : id);
                id += MAXD;
                float l = 0.0f;
                #pragma unroll
                for (int d = 0; d < DH; ++d)
                    l = fmaf(qs[s][h * DH + d], ks[kq][h * DH + d] + srk[id][d], l);
                l *= scale;
                logits[kq] = l;
                mx = fmaxf(mx, l);
            }
            float se = 0.0f;
            #pragma unroll
            for (int kq = 0; kq < SEG; ++kq) {
                logits[kq] = __expf(logits[kq] - mx);
                se += logits[kq];
            }
            float inv = 1.0f / se;
            #pragma unroll
            for (int d = 0; d < DH; ++d) o[h * DH + d] = 0.0f;
            #pragma unroll
            for (int kq = 0; kq < SEG; ++kq) {
                int id = kq - s;
                id = id < -MAXD ? -MAXD : (id > MAXD ? MAXD : id);
                id += MAXD;
                float w = logits[kq] * inv;
                #pragma unroll
                for (int d = 0; d < DH; ++d)
                    o[h * DH + d] = fmaf(w, vs[kq][h * DH + d] + srv[id][d], o[h * DH + d]);
            }
        }
        float a[FREQ], m = 0.0f;
        #pragma unroll
        for (int f = 0; f < FREQ; ++f) { a[f] = o[f] + xs[s][f]; m += a[f]; }
        m *= (1.0f / FREQ);
        float var = 0.0f;
        #pragma unroll
        for (int f = 0; f < FREQ; ++f) { float d = a[f] - m; var = fmaf(d, d, var); }
        float rstd = rsqrtf(var * (1.0f / FREQ) + 1e-3f);
        #pragma unroll
        for (int f = 0; f < FREQ; ++f)
            as[s][f] = lag[f] * (a[f] - m) * rstd + lab[f];

        float ff[FREQ];
        #pragma unroll
        for (int f = 0; f < FREQ; ++f) ff[f] = b2[f];
        for (int j = 0; j < FREQ * 4; ++j) {
            float h1 = b1[j];
            for (int jj = 0; jj < FREQ; ++jj)
                h1 = fmaf(as[s][jj], w1[jj * (FREQ * 4) + j], h1);
            h1 = fmaxf(h1, 0.0f);
            #pragma unroll
            for (int f = 0; f < FREQ; ++f)
                ff[f] = fmaf(h1, w2[j * FREQ + f], ff[f]);
        }
        float m2 = 0.0f;
        #pragma unroll
        for (int f = 0; f < FREQ; ++f) { ff[f] += as[s][f]; m2 += ff[f]; }
        m2 *= (1.0f / FREQ);
        float v2 = 0.0f;
        #pragma unroll
        for (int f = 0; f < FREQ; ++f) { float d = ff[f] - m2; v2 = fmaf(d, d, v2); }
        float r2 = rsqrtf(v2 * (1.0f / FREQ) + 1e-3f);
        #pragma unroll
        for (int f = 0; f < FREQ; ++f) {
            float y = lfg[f] * (ff[f] - m2) * r2 + lfb[f];
            ffbuf[(size_t)n * KPAD + f * SEG + s] = (half_t)y;
        }
    }
    if (s >= SEG && s < SEG + (KPAD - FREQ * SEG))
        ffbuf[(size_t)n * KPAD + FREQ * SEG + (s - SEG)] = (half_t)0.0f;
}

// ---------------------------------------------------------------------------
// Kernel 2: C = relu(A[32768x512]f16 @ B[512x512]f16 + bias), then LN(512).
// Block = 256 threads (8 waves). 32 rows x 512 cols per block.
// A staged to LDS by TDM (double-buffered, TENSORcnt); B direct from L2.
// Per wave: 2 M-tiles x 4 N-tiles -> 8 v_wmma_f32_16x16x32_f16 per K-chunk.
// ---------------------------------------------------------------------------
__global__ __launch_bounds__(256) void dense_relu_ln_kernel(
    const half_t* __restrict__ A,      // [NSEQ][KPAD]
    const half_t* __restrict__ Bw,     // [NUNITS (n-major)][KPAD]
    const float*  __restrict__ bd,
    const float*  __restrict__ g, const float* __restrict__ bb,
    float* __restrict__ out)
{
    __shared__ __align__(128) half_t atile[2][NROWS * 32];   // 2 x 2KB K-chunks
    __shared__ float sdata[NROWS][NUNITS + 8];
    __shared__ float red1[NROWS][8];
    __shared__ float red2[NROWS][8];
    __shared__ float smean[NROWS];
    __shared__ float srstd[NROWS];

    const int tid   = threadIdx.x;
    const int lane  = tid & 31;
    const int wave  = tid >> 5;
    const int r0    = blockIdx.x * NROWS;
    const int mrow  = lane & 15;
    const int khalf = lane >> 4;                     // 0 or 1

    v8f acc[2][4];
    #pragma unroll
    for (int mt = 0; mt < 2; ++mt)
        #pragma unroll
        for (int t = 0; t < 4; ++t)
            acc[mt][t] = (v8f){0, 0, 0, 0, 0, 0, 0, 0};

    const half_t* arowg = A + (size_t)r0 * KPAD;

    // prologue: DMA K-chunk 0 into buffer 0 (wave 0 issues; TDM ignores EXEC)
    if (wave == 0) {
        tdm_load_tile_f16((unsigned)(uintptr_t)&atile[0][0],
                          arowg, 32, NROWS, KPAD);
    }

    for (int kb = 0; kb < KPAD / 32; ++kb) {
        if (wave == 0) {
            if (kb + 1 < KPAD / 32) {
                tdm_load_tile_f16((unsigned)(uintptr_t)&atile[(kb + 1) & 1][0],
                                  arowg + (kb + 1) * 32, 32, NROWS, KPAD);
                __builtin_amdgcn_s_wait_tensorcnt(1);   // oldest (chunk kb) done
            } else {
                __builtin_amdgcn_s_wait_tensorcnt(0);
            }
        }
        __syncthreads();                                // chunk kb visible to all

        // B fragments for this chunk: 4 N-tiles, contiguous 16-K per lane half
        v16h b[4];
        #pragma unroll
        for (int t = 0; t < 4; ++t) {
            int nn = wave * 64 + t * 16 + mrow;
            const half_t* bp = Bw + (size_t)nn * KPAD + kb * 32 + khalf * 16;
            b[t] = *(const v16h*)bp;
        }

        // A fragments from LDS: rows mt*16 + mrow, K contiguous {kh*16..+15}
        v16h a[2];
        #pragma unroll
        for (int mt = 0; mt < 2; ++mt) {
            const half_t* ap = &atile[kb & 1][0] + (mt * 16 + mrow) * 32 + khalf * 16;
            v8h lo = *(const v8h*)(ap);
            v8h hi = *(const v8h*)(ap + 8);
            #pragma unroll
            for (int j = 0; j < 8; ++j) { a[mt][j] = lo[j]; a[mt][8 + j] = hi[j]; }
        }

        #pragma unroll
        for (int mt = 0; mt < 2; ++mt)
            #pragma unroll
            for (int t = 0; t < 4; ++t)
                acc[mt][t] = __builtin_amdgcn_wmma_f32_16x16x32_f16(
                    false, a[mt], false, b[t], (short)0, acc[mt][t], false, false);

        __syncthreads();    // all waves done reading buf[kb&1] before it is re-filled
    }

    // epilogue: bias + relu -> LDS
    #pragma unroll
    for (int mt = 0; mt < 2; ++mt) {
        #pragma unroll
        for (int t = 0; t < 4; ++t) {
            int nn = wave * 64 + t * 16 + mrow;
            float bias = bd[nn];
            #pragma unroll
            for (int v = 0; v < 8; ++v) {
                int m = mt * 16 + v + khalf * 8;     // C/D layout: M = vgpr + 8*(lane>=16)
                float val = acc[mt][t][v] + bias;
                sdata[m][nn] = fmaxf(val, 0.0f);
            }
        }
    }
    __syncthreads();

    // two-stage LN reduction over 512 cols per row (8 partials of 64 per row)
    {
        int row = tid >> 3, seg = tid & 7;
        float s = 0.0f, s2 = 0.0f;
        #pragma unroll 8
        for (int i = 0; i < 64; ++i) {
            float x = sdata[row][seg * 64 + i];
            s += x; s2 = fmaf(x, x, s2);
        }
        red1[row][seg] = s; red2[row][seg] = s2;
    }
    __syncthreads();
    if (tid < NROWS) {
        float ts = 0.0f, ts2 = 0.0f;
        #pragma unroll
        for (int i = 0; i < 8; ++i) { ts += red1[tid][i]; ts2 += red2[tid][i]; }
        float m   = ts * (1.0f / NUNITS);
        float var = ts2 * (1.0f / NUNITS) - m * m;
        smean[tid] = m;
        srstd[tid] = rsqrtf(var + 1e-3f);
    }
    __syncthreads();

    #pragma unroll 4
    for (int i = 0; i < 64; ++i) {
        int flat = i * 256 + tid;                    // 0 .. 16383
        int row  = flat >> 9;
        int col  = flat & 511;
        float x  = sdata[row][col];
        out[(size_t)(r0 + row) * NUNITS + col] =
            g[col] * (x - smean[row]) * srstd[row] + bb[col];
    }
}

// ---------------------------------------------------------------------------
extern "C" void kernel_launch(void* const* d_in, const int* in_sizes, int n_in,
                              void* d_out, int out_size, void* d_ws, size_t ws_size,
                              hipStream_t stream) {
    const float* inp  = (const float*)d_in[0];
    const float* wq   = (const float*)d_in[1];
    const float* bq   = (const float*)d_in[2];
    const float* wk   = (const float*)d_in[3];
    const float* bk   = (const float*)d_in[4];
    const float* wv   = (const float*)d_in[5];
    const float* bv   = (const float*)d_in[6];
    const float* relk = (const float*)d_in[7];
    const float* relv = (const float*)d_in[8];
    const float* lag  = (const float*)d_in[9];
    const float* lab  = (const float*)d_in[10];
    const float* w1   = (const float*)d_in[11];
    const float* b1   = (const float*)d_in[12];
    const float* w2   = (const float*)d_in[13];
    const float* b2   = (const float*)d_in[14];
    const float* lfg  = (const float*)d_in[15];
    const float* lfb  = (const float*)d_in[16];
    const float* wd   = (const float*)d_in[17];
    const float* bd   = (const float*)d_in[18];
    const float* log_ = (const float*)d_in[19];
    const float* lob  = (const float*)d_in[20];

    half_t* ffbuf = (half_t*)d_ws;                                   // 32 MB
    half_t* wdT   = (half_t*)((char*)d_ws + (size_t)NSEQ * KPAD * sizeof(half_t));

    convert_wd_kernel<<<(NUNITS * KPAD) / 256, 256, 0, stream>>>(wd, wdT);

    attn_ff_kernel<<<NSEQ, 32, 0, stream>>>(
        inp, wq, bq, wk, bk, wv, bv, relk, relv, lag, lab,
        w1, b1, w2, b2, lfg, lfb, ffbuf);

    dense_relu_ln_kernel<<<NSEQ / NROWS, 256, 0, stream>>>(
        ffbuf, wdT, bd, log_, lob, (float*)d_out);
}
